// EnhancedEdgeScorer_68753836474931
// MI455X (gfx1250) — matrix operand order repaired and checked by reading, not compiled, not run on hardware
//
#include <hip/hip_runtime.h>

#define N_NODES 32768
#define CTX_L   33
#define N_EDGES 131072

typedef __attribute__((ext_vector_type(16))) __bf16 v16bf;
typedef __attribute__((ext_vector_type(8)))  __bf16 v8bf;
typedef __attribute__((ext_vector_type(8)))  float  v8f;

// ---------------------------------------------------------------------------
// WMMA fragment loaders (CDNA5 16x16x32 bf16 layouts, wave32)
// A (16xK tile, row stride ldk elements):
//   lanes 0-15 : M=lane, elems 0..7 = K kb+0..7,  elems 8..15 = K kb+16..23
//   lanes 16-31: M=lane-16, elems 0..7 = K kb+8..15, elems 8..15 = K kb+24..31
// ---------------------------------------------------------------------------
static __device__ inline v16bf load_A_frag(const __bf16* A, int ldk, int kb, int lane) {
  int m    = lane & 15;
  int koff = kb + ((lane >> 4) << 3);
  const __bf16* p = A + m * ldk + koff;
  v8bf lo = *(const v8bf*)(p);
  v8bf hi = *(const v8bf*)(p + 16);
  return __builtin_shufflevector(lo, hi, 0,1,2,3,4,5,6,7,8,9,10,11,12,13,14,15);
}

// B from torch-convention weight W[Nout x K] (we need B[k][n] = W[n][k]):
//   lanes 0-15 : N=n0+lane,    elems 0..7 = K kb+0..7,  elems 8..15 = K kb+16..23
//   lanes 16-31: N=n0+lane-16, elems 0..7 = K kb+8..15, elems 8..15 = K kb+24..31
static __device__ inline v16bf load_B_frag(const __bf16* W, int ldk, int n0, int kb, int lane) {
  int n    = n0 + (lane & 15);
  int koff = kb + ((lane >> 4) << 3);
  const __bf16* p = W + n * ldk + koff;
  v8bf lo = *(const v8bf*)(p);
  v8bf hi = *(const v8bf*)(p + 16);
  return __builtin_shufflevector(lo, hi, 0,1,2,3,4,5,6,7,8,9,10,11,12,13,14,15);
}

// One 16x16 output tile: C = A[16xK] @ W[n0:n0+16, :K]^T + bias, optional ReLU,
// stored to f32 and/or bf16 destination with leading dim ldout.
template<bool RELU>
static __device__ inline void tile_gemm_store(
    const __bf16* A, int K, const __bf16* W, const float* bias, int colBase,
    float* out_f, __bf16* out_b, size_t ldout, size_t rowBase, int lane) {
  v8f acc = {};
  for (int kb = 0; kb < K; kb += 32) {
    v16bf a = load_A_frag(A, K, kb, lane);
    v16bf b = load_B_frag(W, K, colBase, kb, lane);
    acc = __builtin_amdgcn_wmma_f32_16x16x32_bf16(false, a, false, b,
                                                  (short)0, acc, false, false);
  }
  int n  = colBase + (lane & 15);
  float bv = bias[n];
  int mb = (lane >> 4) << 3;          // C layout: lanes>=16 hold rows 8..15
#pragma unroll
  for (int r = 0; r < 8; r++) {
    float v = acc[r] + bv;
    if (RELU) v = fmaxf(v, 0.f);
    size_t o = (rowBase + (size_t)(mb + r)) * ldout + n;
    if (out_f) out_f[o] = v;
    if (out_b) out_b[o] = (__bf16)v;
  }
}

// ---------------------------------------------------------------------------
// Weight prep: f32 -> bf16 (and W1 zero-padded 258 -> 288 cols)
// ---------------------------------------------------------------------------
__global__ void k_cvt_bf16(const float* __restrict__ src, __bf16* __restrict__ dst, int n) {
  int i = blockIdx.x * blockDim.x + threadIdx.x;
  if (i < n) dst[i] = (__bf16)src[i];
}

__global__ void k_pack_w1(const float* __restrict__ src, __bf16* __restrict__ dst) {
  int i = blockIdx.x * blockDim.x + threadIdx.x;   // over 128*288
  if (i >= 128 * 288) return;
  int r = i / 288, c = i % 288;
  dst[i] = (__bf16)((c < 258) ? src[r * 258 + c] : 0.f);
}

// ---------------------------------------------------------------------------
// Node feature encoding: build [16 x 160] feature tile in LDS, WMMA -> xb
// feats = [type_emb(64) | cat0(32) | cat1(32) | relu(ld*w+b)(32)] @ proj^T + b
// ---------------------------------------------------------------------------
__global__ void k_encode(const int* __restrict__ type_idx, const int* __restrict__ cat_idx,
                         const float* __restrict__ log_deg,
                         const float* __restrict__ type_emb, const float* __restrict__ cat0,
                         const float* __restrict__ cat1,
                         const float* __restrict__ deg_w, const float* __restrict__ deg_b,
                         const __bf16* __restrict__ projW, const float* __restrict__ proj_b,
                         __bf16* __restrict__ xb) {
  __shared__ __align__(16) __bf16 A[16 * 160];
  int t = threadIdx.x;
  size_t tile = blockIdx.x;
  int nit = t >> 4;                         // node within tile (16 threads/node)
  size_t node = tile * 16 + nit;
  int sub = t & 15;
  int ti = type_idx[node];
  int c0 = cat_idx[node * 2 + 0];
  int c1 = cat_idx[node * 2 + 1];
  float ld = log_deg[node];
#pragma unroll
  for (int k = 0; k < 10; k++) {
    int f = sub + k * 16;
    float v;
    if (f < 64)       v = type_emb[ti * 64 + f];
    else if (f < 96)  v = cat0[c0 * 32 + (f - 64)];
    else if (f < 128) v = cat1[c1 * 32 + (f - 96)];
    else              v = fmaxf(ld * deg_w[f - 128] + deg_b[f - 128], 0.f);
    A[nit * 160 + f] = (__bf16)v;
  }
  __syncthreads();
  int wave = t >> 5, lane = t & 31;
  tile_gemm_store<false>(A, 160, projW, proj_b, wave * 16,
                         nullptr, xb, 128, tile * 16, lane);
}

// ---------------------------------------------------------------------------
// QKV GEMM: QKV[N x 384] = xb[N x 128] @ Wqkv[384 x 128]^T + b  (f32 out)
// grid.x = N/16, grid.y = 3 (q/k/v 128-col group), 8 waves x 16-col tiles
// ---------------------------------------------------------------------------
__global__ void k_qkv(const __bf16* __restrict__ xb, const __bf16* __restrict__ W,
                      const float* __restrict__ bias, float* __restrict__ QKV) {
  __shared__ __align__(16) __bf16 A[16 * 128];
  int t = threadIdx.x;
  size_t tile = blockIdx.x;
  ((uint4*)A)[t] = ((const uint4*)(xb + tile * 16 * 128))[t];   // 4 KB stage
  __syncthreads();
  int wave = t >> 5, lane = t & 31;
  int colBase = blockIdx.y * 128 + wave * 16;
  tile_gemm_store<false>(A, 128, W, bias, colBase, QKV, nullptr, 384, tile * 16, lane);
}

// ---------------------------------------------------------------------------
// Fused attention (gathered K/V rows, online softmax, wave32 butterfly
// reductions) + output projection WMMA + ReLU -> new xb
// One wave handles 2 nodes; block of 8 waves = 16-node tile.
// QKV row layout: [q(128) | k(128) | v(128)], head h owns dims h*32..h*32+31.
// ---------------------------------------------------------------------------
__global__ void k_attn(const float* __restrict__ QKV, const int* __restrict__ ctx_idx,
                       const int* __restrict__ pad_mask,
                       const __bf16* __restrict__ outW, const float* __restrict__ out_b,
                       __bf16* __restrict__ xb) {
  __shared__ __align__(16) __bf16 O[16 * 128];
  int t = threadIdx.x, wave = t >> 5, lane = t & 31;
  size_t tile = blockIdx.x;
  for (int s = 0; s < 2; s++) {
    int nit = wave * 2 + s;
    size_t node = tile * 16 + nit;
    const float* qrow = QKV + node * 384;
    float qh[4], mmax[4], den[4], acc[4];
#pragma unroll
    for (int h = 0; h < 4; h++) {
      qh[h] = qrow[h * 32 + lane];
      mmax[h] = -1e30f; den[h] = 0.f; acc[h] = 0.f;
    }
    const int* cp = ctx_idx + node * CTX_L;
    const int* mp = pad_mask + node * CTX_L;
    for (int j = 0; j < CTX_L; j++) {
      if (mp[j]) continue;                       // padded position (wave-uniform)
      size_t c = (size_t)cp[j];
      const float* kr = QKV + c * 384 + 128;
      const float* vr = kr + 128;
      float s4[4];
#pragma unroll
      for (int h = 0; h < 4; h++) s4[h] = qh[h] * kr[h * 32 + lane];
#pragma unroll
      for (int off = 16; off > 0; off >>= 1)
#pragma unroll
        for (int h = 0; h < 4; h++) s4[h] += __shfl_xor(s4[h], off, 32);
#pragma unroll
      for (int h = 0; h < 4; h++) {
        float sc = s4[h] * 0.1767766952966369f;  // 1/sqrt(32)
        float mn = fmaxf(mmax[h], sc);
        float sl = __expf(mmax[h] - mn);
        float p  = __expf(sc - mn);
        den[h] = den[h] * sl + p;
        acc[h] = acc[h] * sl + p * vr[h * 32 + lane];
        mmax[h] = mn;
      }
    }
#pragma unroll
    for (int h = 0; h < 4; h++)
      O[nit * 128 + h * 32 + lane] = (__bf16)(acc[h] / den[h]);
  }
  __syncthreads();
  tile_gemm_store<true>(O, 128, outW, out_b, wave * 16,
                        nullptr, xb, 128, tile * 16, lane);
}

// ---------------------------------------------------------------------------
// Fully fused edge MLP per 16-edge tile:
//   A[16x288] = [xb[u] | xb[v] | feats | 0-pad] (LDS)
//   H = relu(A @ W1p^T + b1)   (WMMA K=288, bf16 in LDS)
//   h2 = relu(H @ W2^T + b2)   (WMMA K=128), logit = h2 . w3 + b3 (LDS atomics)
// ---------------------------------------------------------------------------
__global__ void k_edge(const __bf16* __restrict__ xb, const int* __restrict__ eu,
                       const int* __restrict__ ev, const float* __restrict__ efeat,
                       const __bf16* __restrict__ W1, const float* __restrict__ b1,
                       const __bf16* __restrict__ W2, const float* __restrict__ b2,
                       const float* __restrict__ w3, const float* __restrict__ b3,
                       float* __restrict__ out) {
  __shared__ __align__(16) __bf16 A[16 * 288];
  __shared__ __align__(16) __bf16 H[16 * 128];
  __shared__ float lg[16];
  int t = threadIdx.x, wave = t >> 5, lane = t & 31;
  size_t tile = blockIdx.x;

  // gather xb rows (32 x 16B chunks per edge-row: 16 for u, 16 for v)
  for (int c = t; c < 512; c += 256) {
    int row = c >> 5;
    int q = c & 31;
    size_t node = (size_t)((q < 16) ? eu[tile * 16 + row] : ev[tile * 16 + row]);
    int qq = q & 15;
    ((uint4*)(A + row * 288))[q] = ((const uint4*)(xb + node * 128))[qq];
  }
  if (t < 16) {
    int row = t;
    size_t e = tile * 16 + row;
    A[row * 288 + 256] = (__bf16)efeat[e * 2 + 0];
    A[row * 288 + 257] = (__bf16)efeat[e * 2 + 1];
#pragma unroll
    for (int i = 258; i < 288; i++) A[row * 288 + i] = (__bf16)0.f;
  }
  __syncthreads();

  tile_gemm_store<true>(A, 288, W1, b1, wave * 16, nullptr, H, 128, 0, lane);
  __syncthreads();
  if (t < 16) lg[t] = b3[0];
  __syncthreads();

  if (wave < 4) {                     // 64 output cols of layer 2
    int colBase = wave * 16;
    v8f acc = {};
    for (int kb = 0; kb < 128; kb += 32) {
      v16bf a = load_A_frag(H, 128, kb, lane);
      v16bf b = load_B_frag(W2, 128, colBase, kb, lane);
      acc = __builtin_amdgcn_wmma_f32_16x16x32_bf16(false, a, false, b,
                                                    (short)0, acc, false, false);
    }
    int n = colBase + (lane & 15);
    float bv = b2[n], wv = w3[n];
    int mb = (lane >> 4) << 3;
#pragma unroll
    for (int r = 0; r < 8; r++) {
      float h2 = fmaxf(acc[r] + bv, 0.f);
      atomicAdd(&lg[mb + r], h2 * wv);     // ds_add_f32
    }
  }
  __syncthreads();
  if (t < 16) out[tile * 16 + t] = lg[t];
}

// ---------------------------------------------------------------------------
extern "C" void kernel_launch(void* const* d_in, const int* in_sizes, int n_in,
                              void* d_out, int out_size, void* d_ws, size_t ws_size,
                              hipStream_t stream) {
  (void)in_sizes; (void)n_in; (void)out_size; (void)ws_size;
  const int*   type_idx = (const int*)  d_in[0];
  const int*   cat_idx  = (const int*)  d_in[1];
  const float* log_deg  = (const float*)d_in[2];
  const int*   ctx_idx  = (const int*)  d_in[3];
  const int*   pad_mask = (const int*)  d_in[4];
  const int*   edge_u   = (const int*)  d_in[5];
  const int*   edge_v   = (const int*)  d_in[6];
  const float* edge_f   = (const float*)d_in[7];
  const float* type_emb = (const float*)d_in[8];
  const float* cat0     = (const float*)d_in[9];
  const float* cat1     = (const float*)d_in[10];
  const float* deg_w    = (const float*)d_in[11];
  const float* deg_b    = (const float*)d_in[12];
  const float* proj_w   = (const float*)d_in[13];
  const float* proj_b   = (const float*)d_in[14];
  const float* qkv_w    = (const float*)d_in[15];
  const float* qkv_b    = (const float*)d_in[16];
  const float* out_w    = (const float*)d_in[17];
  const float* out_b    = (const float*)d_in[18];
  const float* w1       = (const float*)d_in[19];
  const float* b1       = (const float*)d_in[20];
  const float* w2       = (const float*)d_in[21];
  const float* b2       = (const float*)d_in[22];
  const float* w3       = (const float*)d_in[23];
  const float* b3       = (const float*)d_in[24];
  float* out = (float*)d_out;

  char* ws = (char*)d_ws;
  size_t off = 0;
  auto carve = [&](size_t bytes) -> char* {
    char* p = ws + off; off += (bytes + 255) & ~(size_t)255; return p;
  };
  __bf16* xb    = (__bf16*)carve((size_t)N_NODES * 128 * 2);   // 8.4 MB
  float*  QKV   = (float*) carve((size_t)N_NODES * 384 * 4);   // 50.3 MB
  __bf16* projW = (__bf16*)carve(128 * 160 * 2);
  __bf16* qkvW  = (__bf16*)carve(3 * 384 * 128 * 2);
  __bf16* outW  = (__bf16*)carve(3 * 128 * 128 * 2);
  __bf16* w1p   = (__bf16*)carve(128 * 288 * 2);
  __bf16* w2b   = (__bf16*)carve(64 * 128 * 2);

  k_cvt_bf16<<<(128 * 160 + 255) / 256, 256, 0, stream>>>(proj_w, projW, 128 * 160);
  k_cvt_bf16<<<(3 * 384 * 128 + 255) / 256, 256, 0, stream>>>(qkv_w, qkvW, 3 * 384 * 128);
  k_cvt_bf16<<<(3 * 128 * 128 + 255) / 256, 256, 0, stream>>>(out_w, outW, 3 * 128 * 128);
  k_cvt_bf16<<<(64 * 128 + 255) / 256, 256, 0, stream>>>(w2, w2b, 64 * 128);
  k_pack_w1<<<(128 * 288 + 255) / 256, 256, 0, stream>>>(w1, w1p);

  k_encode<<<N_NODES / 16, 256, 0, stream>>>(type_idx, cat_idx, log_deg,
      type_emb, cat0, cat1, deg_w, deg_b, projW, proj_b, xb);

  for (int l = 0; l < 3; l++) {
    k_qkv<<<dim3(N_NODES / 16, 3), 256, 0, stream>>>(
        xb, qkvW + (size_t)l * 384 * 128, qkv_b + l * 384, QKV);
    k_attn<<<N_NODES / 16, 256, 0, stream>>>(
        QKV, ctx_idx, pad_mask, outW + (size_t)l * 128 * 128, out_b + l * 128, xb);
  }

  k_edge<<<N_EDGES / 16, 256, 0, stream>>>(xb, edge_u, edge_v, edge_f,
      w1p, b1, w2b, b2, w3, b3, out);
}
